// HiddenLayer_27865747816677
// MI455X (gfx1250) — matrix operand order repaired and checked by reading, not compiled, not run on hardware
//
#include <hip/hip_runtime.h>
#include <hip/hip_bf16.h>

// ---- fixed problem shapes (from reference setup_inputs) ----
#define NT   1508      // Nt
#define QD   12        // Q (= D)
#define LT   8         // Lt
#define DIN  192       // 2*Lt*Q
#define MM   192       // M (inducing points)
#define NR   1500      // Nr = Nt - Lt  (hankel rows)
#define NRP  1504      // Nr padded to multiple of 16

typedef __attribute__((ext_vector_type(16))) _Float16 v16h;
typedef __attribute__((ext_vector_type(8)))  _Float16 v8h;
typedef __attribute__((ext_vector_type(8)))  float    v8f;

// ---------------- WMMA fragment helpers (CDNA5 16x16x32 f16) ----------------
// A-matrix 16x32 f16 (row-major source, lda in elements):
//   lane L: row = L&15; half = L>>4; elems 0..15 = A[row, k0+half*16 .. +15]
//   (contiguous k, matching load_b_frag)
__device__ __forceinline__ v16h load_a_frag(const _Float16* A, int lda, int row0, int k0) {
    int lane = threadIdx.x & 31;
    const _Float16* p = A + (size_t)(row0 + (lane & 15)) * lda + k0 + ((lane >> 4) << 4);
    v8h lo = *reinterpret_cast<const v8h*>(p);
    v8h hi = *reinterpret_cast<const v8h*>(p + 8);
    return __builtin_shufflevector(lo, hi, 0,1,2,3,4,5,6,7,8,9,10,11,12,13,14,15);
}
// B-matrix 32x16 f16 given Bt = B^T row-major (N x K, ldb in elements):
//   lane L: col = L&15; half = L>>4; elems = B[k0+half*16 .. +15, col] contiguous in Bt.
__device__ __forceinline__ v16h load_b_frag(const _Float16* Bt, int ldb, int col0, int k0) {
    int lane = threadIdx.x & 31;
    const _Float16* p = Bt + (size_t)(col0 + (lane & 15)) * ldb + k0 + ((lane >> 4) << 4);
    return *reinterpret_cast<const v16h*>(p);
}

// ---------------- Z prep: f16 copies of Z and Z^2 ----------------
__global__ void zprep_kernel(const float* __restrict__ Z, _Float16* __restrict__ Zh,
                             _Float16* __restrict__ Z2h) {
    int idx = blockIdx.x * blockDim.x + threadIdx.x;
    if (idx < MM * DIN) {
        float z = Z[idx];
        Zh[idx]  = (_Float16)z;
        Z2h[idx] = (_Float16)(z * z);
    }
}

// ---------------- d2zz + Kuu ----------------
__global__ void d2zz_kernel(const float* __restrict__ Z, const float* __restrict__ ls,
                            const float* __restrict__ kern_var,
                            float* __restrict__ d2zz, float* __restrict__ Kuu) {
    int m = blockIdx.x, p = threadIdx.x;   // grid MM, block MM
    float s = 0.f;
    for (int q = 0; q < DIN; ++q) {
        float inv = 1.f / ls[q];
        float d = (Z[(size_t)m * DIN + q] - Z[(size_t)p * DIN + q]) * inv;
        s += d * d;
    }
    d2zz[(size_t)m * MM + p] = s;
    float kv = kern_var[0];
    Kuu[(size_t)m * MM + p] = kv * __expf(-0.5f * s) + ((m == p) ? 1e-6f : 0.f);
}

// ---------------- per-row prep: hankel gather + derived operands ----------------
__global__ __launch_bounds__(256) void prep_kernel(
        const float* __restrict__ Xm_m, const float* __restrict__ Xm_v,
        const float* __restrict__ X_mean, const float* __restrict__ X_var,
        const float* __restrict__ ls,
        _Float16* __restrict__ A1h, _Float16* __restrict__ I1h,
        _Float16* __restrict__ A2h, _Float16* __restrict__ W2h,
        float* __restrict__ s1, float* __restrict__ coef) {
    int n = blockIdx.x, tid = threadIdx.x;      // grid NRP, block 256 (192 active cols)
    __shared__ float4 red[256];
    float t1 = 0.f, t2 = 0.f, t3 = 0.f, t4 = 0.f;
    if (tid < DIN) {
        size_t o = (size_t)n * DIN + tid;
        if (n < NR) {
            int q = tid; float xm, xv;
            if (q < LT * QD) {                 // first hankel half: X_mean / X_var
                int j = q / QD, qq = q % QD;
                xm = X_mean[(size_t)(n + j) * QD + qq];
                xv = X_var [(size_t)(n + j) * QD + qq];
            } else {                           // second half: Xm_m / Xm_v shifted by 1
                int q2 = q - LT * QD, j = q2 / QD, qq = q2 % QD;
                xm = Xm_m[(size_t)(1 + n + j) * QD + qq];
                xv = Xm_v[(size_t)(1 + n + j) * QD + qq];
            }
            float l = ls[q], l2 = l * l;
            float d1 = l2 + xv, d2 = l2 + 2.f * xv;
            A1h[o] = (_Float16)(xm / d1);  I1h[o] = (_Float16)(1.f / d1);
            A2h[o] = (_Float16)(xm / d2);  W2h[o] = (_Float16)(1.f / d2);
            t1 = log1pf(xv / l2);  t2 = xm * xm / d1;
            t3 = log1pf(2.f * xv / l2);  t4 = xm * xm / d2;
        } else {                               // zero-pad rows
            A1h[o] = (_Float16)0.f; I1h[o] = (_Float16)0.f;
            A2h[o] = (_Float16)0.f; W2h[o] = (_Float16)0.f;
        }
    }
    red[tid] = make_float4(t1, t2, t3, t4);
    __syncthreads();
    for (int s = 128; s > 0; s >>= 1) {
        if (tid < s) {
            float4 a = red[tid], b = red[tid + s];
            red[tid] = make_float4(a.x + b.x, a.y + b.y, a.z + b.z, a.w + b.w);
        }
        __syncthreads();
    }
    if (tid == 0) {
        float4 r = red[0];
        if (n < NR) { s1[n] = -0.5f * r.x - 0.5f * r.y;  coef[n] = -0.5f * r.z - r.w; }
        else        { s1[n] = -1e30f;                    coef[n] = -1e30f; }
    }
}

// ---------------- dual WMMA GEMM: O0 = A0 @ B0t^T, O1 = A1 @ B1t^T ----------------
__global__ __launch_bounds__(32) void gemm2_wmma(
        const _Float16* __restrict__ A0, const _Float16* __restrict__ A1v,
        const _Float16* __restrict__ B0t, const _Float16* __restrict__ B1t,
        float* __restrict__ O0, float* __restrict__ O1) {
    int row0 = blockIdx.x * 16, col0 = blockIdx.y * 16;
    int lane = threadIdx.x & 31;
    v8f acc0 = {}; v8f acc1 = {};
    #pragma unroll
    for (int k0 = 0; k0 < DIN; k0 += 32) {
        v16h a0 = load_a_frag(A0, DIN, row0, k0);
        v16h b0 = load_b_frag(B0t, DIN, col0, k0);
        acc0 = __builtin_amdgcn_wmma_f32_16x16x32_f16(false, a0, false, b0, (short)0, acc0, false, false);
        v16h a1 = load_a_frag(A1v, DIN, row0, k0);
        v16h b1 = load_b_frag(B1t, DIN, col0, k0);
        acc1 = __builtin_amdgcn_wmma_f32_16x16x32_f16(false, a1, false, b1, (short)0, acc1, false, false);
    }
    int half = lane >> 4, cn = lane & 15;
    #pragma unroll
    for (int r = 0; r < 8; ++r) {
        int row = row0 + r + 8 * half;
        O0[(size_t)row * MM + col0 + cn] = acc0[r];
        O1[(size_t)row * MM + col0 + cn] = acc1[r];
    }
}

// ---------------- psi1 epilogue ----------------
__global__ void psi1_finish(const float* __restrict__ Pq, const float* __restrict__ Cq,
                            const float* __restrict__ s1, const float* __restrict__ kern_var,
                            float* __restrict__ psi1) {
    int idx = blockIdx.x * blockDim.x + threadIdx.x;
    if (idx >= NRP * MM) return;
    int n = idx / MM;
    psi1[idx] = kern_var[0] * __expf(s1[n] + Pq[idx] - 0.5f * Cq[idx]);
}

// ---------------- fused psi2: e-GEMM (WMMA) + exp + n-reduction ----------------
// grid (12,12): 16x16 (m,p) tile per block, block 256 = 8 waves (2 col-tiles each)
__global__ __launch_bounds__(256) void psi2_fused(
        const float* __restrict__ Z, const float* __restrict__ d2zz,
        const float* __restrict__ bmat, const float* __restrict__ cmat,
        const float* __restrict__ coef, const _Float16* __restrict__ W2h,
        const float* __restrict__ kern_var, float* __restrict__ psi2) {
    extern __shared__ float dynsm[];
    _Float16* Gs = reinterpret_cast<_Float16*>(dynsm);   // 256 * 192 f16 (Bt layout, ldb=DIN)
    float* accs = dynsm + (256 * DIN) / 2;               // 256 f32
    int m0 = blockIdx.x * 16, p0 = blockIdx.y * 16;
    int tid = threadIdx.x;

    for (int idx = tid; idx < 256 * DIN; idx += 256) {
        int c = idx / DIN, q = idx % DIN;
        int mi = c >> 4, pi = c & 15;
        Gs[(size_t)c * DIN + q] =
            (_Float16)(Z[(size_t)(m0 + mi) * DIN + q] * Z[(size_t)(p0 + pi) * DIN + q]);
    }
    accs[tid] = 0.f;
    __syncthreads();

    int wave = tid >> 5, lane = tid & 31, half = lane >> 4, pi = lane & 15;
    int ct0 = wave * 2;
    int m_a = m0 + ct0, m_b = m0 + ct0 + 1, p = p0 + pi;
    float dz_a = -0.25f * d2zz[(size_t)m_a * MM + p];
    float dz_b = -0.25f * d2zz[(size_t)m_b * MM + p];
    float part_a = 0.f, part_b = 0.f;

    for (int n0 = 0; n0 < NRP; n0 += 16) {
        v16h af[6];
        #pragma unroll
        for (int kk = 0; kk < 6; ++kk) af[kk] = load_a_frag(W2h, DIN, n0, kk * 32);
        v8f ea = {}, eb = {};
        #pragma unroll
        for (int kk = 0; kk < 6; ++kk) {
            v16h bfa = load_b_frag(Gs, DIN, ct0 * 16, kk * 32);
            ea = __builtin_amdgcn_wmma_f32_16x16x32_f16(false, af[kk], false, bfa, (short)0, ea, false, false);
            v16h bfb = load_b_frag(Gs, DIN, ct0 * 16 + 16, kk * 32);
            eb = __builtin_amdgcn_wmma_f32_16x16x32_f16(false, af[kk], false, bfb, (short)0, eb, false, false);
        }
        #pragma unroll
        for (int r = 0; r < 8; ++r) {
            int n = n0 + r + 8 * half;
            float bp = bmat[(size_t)n * MM + p];
            float cp = cmat[(size_t)n * MM + p];
            float base = coef[n] + bp - 0.25f * cp;
            part_a += __expf(base + dz_a + bmat[(size_t)n * MM + m_a]
                             - 0.25f * cmat[(size_t)n * MM + m_a] - 0.5f * ea[r]);
            part_b += __expf(base + dz_b + bmat[(size_t)n * MM + m_b]
                             - 0.25f * cmat[(size_t)n * MM + m_b] - 0.5f * eb[r]);
        }
    }
    atomicAdd(&accs[ct0 * 16 + pi], part_a);
    atomicAdd(&accs[ct0 * 16 + 16 + pi], part_b);
    __syncthreads();
    float kv = kern_var[0];
    int mi = tid >> 4, pii = tid & 15;
    psi2[(size_t)(m0 + mi) * MM + p0 + pii] = kv * kv * accs[tid];
}

// ---------------- single-block Cholesky in LDS (M=192) ----------------
__global__ void cholesky_kernel(const float* __restrict__ Ain, float* __restrict__ Lout, int addEye) {
    extern __shared__ float dynsm[];
    int tid = threadIdx.x;                      // blockDim 192, thread = row
    for (int idx = tid; idx < MM * MM; idx += blockDim.x) {
        float v = Ain[idx];
        if (addEye && (idx / MM) == (idx % MM)) v += 1.f;
        dynsm[idx] = v;
    }
    __syncthreads();
    for (int k = 0; k < MM; ++k) {
        if (tid == k) dynsm[k * MM + k] = sqrtf(dynsm[k * MM + k]);
        __syncthreads();
        if (tid > k) dynsm[tid * MM + k] /= dynsm[k * MM + k];
        __syncthreads();
        if (tid > k) {
            float lik = dynsm[tid * MM + k];
            for (int j = k + 1; j <= tid; ++j) dynsm[tid * MM + j] -= lik * dynsm[j * MM + k];
        }
        __syncthreads();
    }
    for (int idx = tid; idx < MM * MM; idx += blockDim.x) {
        int i = idx / MM, j = idx % MM;
        Lout[idx] = (j <= i) ? dynsm[idx] : 0.f;
    }
}

// ---------------- lower-triangular solve, one thread per column ----------------
// scaleMode: 0 -> 1.0, 1 -> 1/sqrt(lik_var), 2 -> 1/lik_var
__global__ void trsolve_kernel(const float* __restrict__ L, const float* __restrict__ Bin,
                               int ncols, int ldin, int transIn,
                               const float* __restrict__ lik_var, int scaleMode,
                               float* __restrict__ Xout, int ldout) {
    int col = blockIdx.x * blockDim.x + threadIdx.x;
    if (col >= ncols) return;
    float x[MM];
    for (int i = 0; i < MM; ++i) {
        float s = transIn ? Bin[(size_t)col * ldin + i] : Bin[(size_t)i * ldin + col];
        for (int j = 0; j < i; ++j) s -= L[(size_t)i * MM + j] * x[j];
        x[i] = s / L[(size_t)i * MM + i];
    }
    float scale = 1.f;
    if (scaleMode == 1)      scale = rsqrtf(lik_var[0]);
    else if (scaleMode == 2) scale = 1.f / lik_var[0];
    for (int i = 0; i < MM; ++i) Xout[(size_t)i * ldout + col] = x[i] * scale;
}

// ---------------- AX = A @ X_mo (192 x 12) ----------------
__global__ void ax_kernel(const float* __restrict__ Amat, const float* __restrict__ X_mean,
                          float* __restrict__ AX) {
    int d = blockIdx.x, i = threadIdx.x;        // grid QD, block MM
    float s = 0.f;
    for (int n = 0; n < NR; ++n)
        s += Amat[(size_t)i * NR + n] * X_mean[(size_t)(LT + n) * QD + d];
    AX[i * QD + d] = s;
}

// ---------------- final scalar assembly ----------------
__global__ __launch_bounds__(256) void final_kernel(
        const float* __restrict__ X_mean, const float* __restrict__ X_var,
        const float* __restrict__ AAT, const float* __restrict__ LB,
        const float* __restrict__ cvec, const float* __restrict__ kern_var,
        const float* __restrict__ lik_var, float* __restrict__ out) {
    __shared__ float red[256];
    int tid = threadIdx.x;
    float sLogV = 0.f, sV = 0.f, sM2 = 0.f, sB = 0.f, sTr = 0.f, sLd = 0.f, sC2 = 0.f;
    for (int idx = tid; idx < NR * QD; idx += 256) {     // rows Lt..Nt-1
        int n = idx / QD, d = idx % QD;
        float v = X_var [(size_t)(LT + n) * QD + d];
        float m = X_mean[(size_t)(LT + n) * QD + d];
        sLogV += logf(v); sV += v; sM2 += m * m;
    }
    for (int idx = tid; idx < LT * QD; idx += 256) {     // first Lt rows
        float m = X_mean[idx], v = X_var[idx];
        sB += m * m + v;
    }
    for (int i = tid; i < MM; i += 256) {
        sTr += AAT[(size_t)i * MM + i];
        sLd += logf(LB[(size_t)i * MM + i]);
    }
    for (int i = tid; i < MM * QD; i += 256) { float c = cvec[i]; sC2 += c * c; }

    float vals[7] = { sLogV, sV, sM2, sB, sTr, sLd, sC2 };
    float tot[7];
    for (int v = 0; v < 7; ++v) {
        red[tid] = vals[v]; __syncthreads();
        for (int s = 128; s > 0; s >>= 1) { if (tid < s) red[tid] += red[tid + s]; __syncthreads(); }
        tot[v] = red[0]; __syncthreads();
    }
    if (tid == 0) {
        float sigma2 = lik_var[0], kv = kern_var[0];
        const float lp2pi = 1.8378770664093453f;
        float D = (float)QD, ND = (float)(NR * QD);
        float psi0 = kv * (float)NR;
        float logdetB = 2.f * tot[5];
        float bound = -0.5f * ND * (lp2pi + logf(sigma2));
        bound += -0.5f / sigma2 * (tot[1] + tot[2]);
        bound += -0.5f * D * (psi0 / sigma2 - tot[4]);
        bound += -0.5f * D * logdetB;
        bound += 0.5f * tot[6];
        bound += 0.5f * tot[0] + ND * 0.5f * lp2pi;                  // ent
        bound += -(float)(LT * QD) * lp2pi - 0.5f * tot[3];          // ent2
        out[0] = bound;
    }
}

extern "C" void kernel_launch(void* const* d_in, const int* in_sizes, int n_in,
                              void* d_out, int out_size, void* d_ws, size_t ws_size,
                              hipStream_t stream) {
    const float* Xm_m   = (const float*)d_in[1];
    const float* Xm_v   = (const float*)d_in[2];
    const float* Zp     = (const float*)d_in[3];
    const float* X_mean = (const float*)d_in[4];
    const float* X_var  = (const float*)d_in[5];
    const float* kv     = (const float*)d_in[6];
    const float* ls     = (const float*)d_in[7];
    const float* lv     = (const float*)d_in[8];
    float* outp = (float*)d_out;

    char* base = (char*)d_ws;
    size_t off = 0;
    auto alloc = [&](size_t bytes) { size_t o = off; off = (off + bytes + 255) & ~(size_t)255; return o; };

    _Float16* A1h = (_Float16*)(base + alloc((size_t)NRP * DIN * 2));
    _Float16* I1h = (_Float16*)(base + alloc((size_t)NRP * DIN * 2));
    _Float16* A2h = (_Float16*)(base + alloc((size_t)NRP * DIN * 2));
    _Float16* W2h = (_Float16*)(base + alloc((size_t)NRP * DIN * 2));
    _Float16* Zh  = (_Float16*)(base + alloc((size_t)MM * DIN * 2));
    _Float16* Z2h = (_Float16*)(base + alloc((size_t)MM * DIN * 2));
    float* s1   = (float*)(base + alloc((size_t)NRP * 4));
    float* coef = (float*)(base + alloc((size_t)NRP * 4));
    float* d2zz = (float*)(base + alloc((size_t)MM * MM * 4));
    float* Kuu  = (float*)(base + alloc((size_t)MM * MM * 4));
    float* Lc   = (float*)(base + alloc((size_t)MM * MM * 4));
    float* Pq   = (float*)(base + alloc((size_t)NRP * MM * 4));   // reused as bmat
    float* Cq   = (float*)(base + alloc((size_t)NRP * MM * 4));   // reused as cmat
    float* psi1 = (float*)(base + alloc((size_t)NRP * MM * 4));
    float* psi2 = (float*)(base + alloc((size_t)MM * MM * 4));
    float* tmpm = (float*)(base + alloc((size_t)MM * MM * 4));
    float* AAT  = (float*)(base + alloc((size_t)MM * MM * 4));
    float* LB   = (float*)(base + alloc((size_t)MM * MM * 4));
    float* Amat = (float*)(base + alloc((size_t)MM * NR * 4));
    float* AX   = (float*)(base + alloc((size_t)MM * QD * 4));
    float* cvec = (float*)(base + alloc((size_t)MM * QD * 4));
    (void)ws_size; (void)in_sizes; (void)n_in; (void)out_size;

    // 1) operand prep
    zprep_kernel<<<(MM * DIN + 255) / 256, 256, 0, stream>>>(Zp, Zh, Z2h);
    d2zz_kernel<<<MM, MM, 0, stream>>>(Zp, ls, kv, d2zz, Kuu);
    prep_kernel<<<NRP, 256, 0, stream>>>(Xm_m, Xm_v, X_mean, X_var, ls,
                                         A1h, I1h, A2h, W2h, s1, coef);
    // 2) psi1 GEMMs + epilogue
    gemm2_wmma<<<dim3(NRP / 16, MM / 16), 32, 0, stream>>>(A1h, I1h, Zh, Z2h, Pq, Cq);
    psi1_finish<<<(NRP * MM + 255) / 256, 256, 0, stream>>>(Pq, Cq, s1, kv, psi1);
    // 3) b / cmat GEMMs (reuse Pq/Cq buffers)
    gemm2_wmma<<<dim3(NRP / 16, MM / 16), 32, 0, stream>>>(A2h, W2h, Zh, Z2h, Pq, Cq);
    // 4) fused psi2 (e-GEMM via WMMA + exp + n-reduction), 96KB f16 G-tile + 1KB acc in LDS
    size_t psi2_lds = (size_t)256 * DIN * 2 + 256 * 4;
    psi2_fused<<<dim3(MM / 16, MM / 16), 256, psi2_lds, stream>>>(Zp, d2zz, Pq, Cq, coef,
                                                                  W2h, kv, psi2);
    // 5) Cholesky(Kuu) and solves
    size_t chol_lds = (size_t)MM * MM * 4;
    cholesky_kernel<<<1, MM, chol_lds, stream>>>(Kuu, Lc, 0);
    trsolve_kernel<<<(NR + 127) / 128, 128, 0, stream>>>(Lc, psi1, NR, MM, 1, lv, 1, Amat, NR);
    trsolve_kernel<<<1, MM, 0, stream>>>(Lc, psi2, MM, MM, 0, lv, 0, tmpm, MM);
    trsolve_kernel<<<1, MM, 0, stream>>>(Lc, tmpm, MM, MM, 1, lv, 2, AAT, MM);
    // 6) B = AAT + I, LB = chol(B)
    cholesky_kernel<<<1, MM, chol_lds, stream>>>(AAT, LB, 1);
    // 7) c = solve(LB, A @ X_mo) / sigma
    ax_kernel<<<QD, MM, 0, stream>>>(Amat, X_mean, AX);
    trsolve_kernel<<<1, 32, 0, stream>>>(LB, AX, QD, QD, 0, lv, 1, cvec, QD);
    // 8) scalar bound
    final_kernel<<<1, 256, 0, stream>>>(X_mean, X_var, AAT, LB, cvec, kv, lv, outp);
}
